// INLAttentionBlock_3599182594277
// MI455X (gfx1250) — matrix operand order repaired and checked by reading, not compiled, not run on hardware
//
#include <hip/hip_runtime.h>
#include <hip/hip_bf16.h>

// ---------------------------------------------------------------------------
// INL attention block for MI455X (gfx1250, wave32, WMMA).
// Mixed precision: f16 matrix operands, f32 accumulation (v_wmma_f32_16x16x32_f16).
// All GEMMs are TN-form (C[m][n] = sum_k A[m][k] * BT[n][k]) so every
// fragment load is a contiguous 32-byte run per lane -> global_load_b128.
// ---------------------------------------------------------------------------

typedef __attribute__((ext_vector_type(16))) _Float16 v16h;
typedef __attribute__((ext_vector_type(8)))  float    v8f;

static constexpr int Bc = 8, Cc = 512, Sc = 1024, NHc = 8, HDc = 64;

__device__ __forceinline__ v8f wmma_f16(v16h a, v16h b, v8f c) {
  // D = A(16x32) * B(32x16) + C(16x16 f32)
  return __builtin_amdgcn_wmma_f32_16x16x32_f16(false, a, false, b, (short)0, c,
                                                false, false);
}

// A fragment: 16x32 f16 (MxK), row-major, stride lda (halves).
// lane m = lane&15 ; half t -> K = t + (lane<16 ? 0 : 16)
__device__ __forceinline__ v16h load_a_frag(const _Float16* A, int lda, int m,
                                            int k0, int lane) {
  const int off = (lane & 16) ? 16 : 0;
  const _Float16* p = A + (size_t)m * lda + k0 + off;
  v16h f;
#pragma unroll
  for (int t = 0; t < 16; ++t) f[t] = p[t];
  return f;
}

// B fragment from TRANSPOSED storage BT[n][k], stride ldk (halves).
// B-fragment layout: lane n = lane&15 ; half t -> K = k0 + t + (lane<16?0:16)
// => 16 contiguous halves (32B) per lane.
__device__ __forceinline__ v16h load_bt_frag(const _Float16* BT, int ldk, int n,
                                             int k0, int lane) {
  const _Float16* p = BT + (size_t)n * ldk + k0 + ((lane & 16) ? 16 : 0);
  v16h f;
#pragma unroll
  for (int t = 0; t < 16; ++t) f[t] = p[t];
  return f;
}

// ---------------------------------------------------------------------------
// Weight prep: fp32 -> f16, natural [o][c] layouts (qkv as A, inl/proj as BT operands).
// ---------------------------------------------------------------------------
__global__ void prep_w_kernel(const float* __restrict__ qkv_w,
                              const float* __restrict__ inl_w,
                              const float* __restrict__ proj_w,
                              _Float16* __restrict__ qkvw16,
                              _Float16* __restrict__ inlw16,
                              _Float16* __restrict__ projw16) {
  int i = blockIdx.x * blockDim.x + threadIdx.x;
  if (i < 3 * Cc * Cc) qkvw16[i] = (_Float16)qkv_w[i];
  if (i < Cc * Cc) {
    inlw16[i]  = (_Float16)inl_w[i];
    projw16[i] = (_Float16)proj_w[i];
  }
}

// ---------------------------------------------------------------------------
// GroupNorm(8, 512) + transpose: x[b][c][s] -> hT[b][s][c] (f16).
// One workgroup per (b, group): 64 ch x 1024 sp. LDS 64x64 tile transpose so
// both the x reads and the hT writes are coalesced.
// ---------------------------------------------------------------------------
__global__ void gn_kernel(const float* __restrict__ x,
                          const float* __restrict__ gns,
                          const float* __restrict__ gnb,
                          _Float16* __restrict__ hT) {
  __shared__ float red[16];
  __shared__ _Float16 tile[64][65];

  const int bg = blockIdx.x;            // 0..63
  const int b = bg >> 3, g = bg & 7;
  const float* xp = x + (((size_t)b * Cc) + g * 64) * Sc;
  const int tid = threadIdx.x;

  float s = 0.f, ss = 0.f;
  for (int i = tid; i < 64 * Sc; i += blockDim.x) {
    float v = xp[i]; s += v; ss += v * v;
  }
#pragma unroll
  for (int m = 16; m >= 1; m >>= 1) {
    s  += __shfl_xor(s,  m, 32);
    ss += __shfl_xor(ss, m, 32);
  }
  const int w = tid >> 5;
  if ((tid & 31) == 0) { red[w] = s; red[w + 8] = ss; }
  __syncthreads();
  if (tid == 0) {
    float ts = 0.f, tss = 0.f;
    for (int i = 0; i < 8; ++i) { ts += red[i]; tss += red[i + 8]; }
    float mean = ts * (1.f / 65536.f);
    float var  = tss * (1.f / 65536.f) - mean * mean;
    red[0] = mean;
    red[1] = rsqrtf(var + 1e-5f);
  }
  __syncthreads();
  const float mean = red[0], inv = red[1];

  for (int st = 0; st < 16; ++st) {       // 16 s-tiles of 64
#pragma unroll
    for (int r = 0; r < 16; ++r) {        // normalize + stage (coalesced reads)
      int c_l = r * 4 + (tid >> 6);
      int s_l = tid & 63;
      int c = g * 64 + c_l;
      float v = xp[(size_t)c_l * Sc + st * 64 + s_l];
      tile[c_l][s_l] = (_Float16)((v - mean) * inv * gns[c] + gnb[c]);
    }
    __syncthreads();
#pragma unroll
    for (int r = 0; r < 16; ++r) {        // transposed write (coalesced)
      int s_l = r * 4 + (tid >> 6);
      int c_l = tid & 63;
      hT[(((size_t)b * Sc) + st * 64 + s_l) * Cc + g * 64 + c_l] = tile[c_l][s_l];
    }
    __syncthreads();
  }
}

// ---------------------------------------------------------------------------
// QKV GEMM per batch: out[o][s] = sum_c W[o][c] * hT[s][c]
// M=1536(o), N=1024(s), K=512.  Wave: 32x32 macro-tile (2x2 blocking).
// Epilogue stages tile in LDS; q,k stored [bh][s][d] (q pre-scaled hd^-0.5),
// v stored [bh][d][s] -- exactly the layouts attention consumes as A / BT.
// ---------------------------------------------------------------------------
__global__ void qkv_gemm_kernel(const _Float16* __restrict__ W,
                                const _Float16* __restrict__ HT,
                                const float* __restrict__ qkv_b,
                                _Float16* __restrict__ q16,
                                _Float16* __restrict__ k16,
                                _Float16* __restrict__ v16) {
  __shared__ float stage[8][32][33];
  const int lane = threadIdx.x & 31;
  const int lm = lane & 15;
  const int w = threadIdx.x >> 5;
  const int task = blockIdx.x * 8 + w;       // < 8*48*32 = 12288
  const int b  = task / (48 * 32);
  const int rm = task % (48 * 32);
  const int m0 = (rm >> 5) * 32, n0 = (rm & 31) * 32;
  const _Float16* Hb = HT + (size_t)b * Sc * Cc;

  v8f acc[2][2] = {};
  for (int k0 = 0; k0 < Cc; k0 += 32) {
    if (k0 + 32 < Cc) {
      __builtin_prefetch(W + (size_t)(m0 + lm) * Cc + k0 + 32, 0, 1);
      __builtin_prefetch(Hb + (size_t)(n0 + lm) * Cc + k0 + 32, 0, 1);
    }
    v16h a0 = load_a_frag(W, Cc, m0 + lm, k0, lane);
    v16h a1 = load_a_frag(W, Cc, m0 + 16 + lm, k0, lane);
    v16h b0 = load_bt_frag(Hb, Cc, n0 + lm, k0, lane);
    v16h b1 = load_bt_frag(Hb, Cc, n0 + 16 + lm, k0, lane);
    acc[0][0] = wmma_f16(a0, b0, acc[0][0]);
    acc[0][1] = wmma_f16(a0, b1, acc[0][1]);
    acc[1][0] = wmma_f16(a1, b0, acc[1][0]);
    acc[1][1] = wmma_f16(a1, b1, acc[1][1]);
  }
#pragma unroll
  for (int im = 0; im < 2; ++im)
#pragma unroll
    for (int in = 0; in < 2; ++in)
#pragma unroll
      for (int r = 0; r < 8; ++r)
        stage[w][im * 16 + r + ((lane & 16) ? 8 : 0)][in * 16 + lm] =
            acc[im][in][r];
  __syncthreads();

  if (m0 < Cc) {                                    // ---- Q: [bh][s][d], *0.125
    const int bh = b * NHc + (m0 >> 6);
    const int dbase = m0 & 63;
    const float bias = qkv_b[m0 + lane];
#pragma unroll 4
    for (int s_l = 0; s_l < 32; ++s_l)
      q16[((size_t)bh * Sc + n0 + s_l) * HDc + dbase + lane] =
          (_Float16)((stage[w][lane][s_l] + bias) * 0.125f);
  } else if (m0 < 2 * Cc) {                         // ---- K: [bh][t][d]
    const int oo = m0 - Cc;
    const int bh = b * NHc + (oo >> 6);
    const int dbase = oo & 63;
    const float bias = qkv_b[m0 + lane];
#pragma unroll 4
    for (int s_l = 0; s_l < 32; ++s_l)
      k16[((size_t)bh * Sc + n0 + s_l) * HDc + dbase + lane] =
          (_Float16)(stage[w][lane][s_l] + bias);
  } else {                                          // ---- V: [bh][d][t]
    const int oo = m0 - 2 * Cc;
    const int bh = b * NHc + (oo >> 6);
    const int dbase = oo & 63;
#pragma unroll 4
    for (int orow = 0; orow < 32; ++orow) {
      const float bias = qkv_b[m0 + orow];
      v16[((size_t)bh * HDc + dbase + orow) * Sc + n0 + lane] =
          (_Float16)(stage[w][orow][lane] + bias);
    }
  }
}

// ---------------------------------------------------------------------------
// Attention: one workgroup (8 waves) per (b, head, 16-row s-tile).
// Phase1: S = Q*K^T (A=q[s][d], BT=k[t][d]); f16 scores in LDS.
// Phase2: row softmax (exp stored, rowsum deferred).
// Phase3: O = P*V (A=probs from LDS, BT=v[d][t]).
// Phase4: combine partials, scale by 1/rowsum, emit hf[b][s][c] (f32 + f16).
// ---------------------------------------------------------------------------
__global__ void attn_kernel(const _Float16* __restrict__ q16,
                            const _Float16* __restrict__ k16,
                            const _Float16* __restrict__ v16,
                            float* __restrict__ hf32,
                            _Float16* __restrict__ hf16) {
  __shared__ _Float16 sc[16 * Sc];    // 32 KB scores/probs
  __shared__ float obuf[8 * 256];     // 8 KB partial O tiles
  __shared__ float rsum[16];

  const int lane = threadIdx.x & 31;
  const int lm = lane & 15;
  const int w = threadIdx.x >> 5;
  const int bh    = blockIdx.x >> 6;  // 0..63
  const int stile = blockIdx.x & 63;  // 0..63
  const int s0 = stile * 16;
  const _Float16* qp = q16 + (size_t)bh * Sc * HDc;   // [s][d]
  const _Float16* kp = k16 + (size_t)bh * Sc * HDc;   // [t][d]
  const _Float16* vp = v16 + (size_t)bh * HDc * Sc;   // [d][t]

  // Phase 1: scores; 8 column-tiles of 16 per wave.
  v16h aq0 = load_a_frag(qp, HDc, s0 + lm, 0, lane);
  v16h aq1 = load_a_frag(qp, HDc, s0 + lm, 32, lane);
  for (int i = 0; i < 8; ++i) {
    const int n0 = (w * 8 + i) * 16;
    v16h b0 = load_bt_frag(kp, HDc, n0 + lm, 0, lane);
    v16h b1 = load_bt_frag(kp, HDc, n0 + lm, 32, lane);
    v8f acc = {};
    acc = wmma_f16(aq0, b0, acc);
    acc = wmma_f16(aq1, b1, acc);
#pragma unroll
    for (int r = 0; r < 8; ++r) {
      int m = r + ((lane & 16) ? 8 : 0);
      sc[m * Sc + n0 + lm] = (_Float16)acc[r];
    }
  }
  __syncthreads();

  // Phase 2: softmax per row; 16 contiguous lanes per row -> shfl_xor reduce.
  {
    const int row = threadIdx.x >> 4;
    const int j   = threadIdx.x & 15;
    _Float16* rp = sc + row * Sc;
    float mx = -1e30f;
    for (int p = j; p < Sc; p += 16) mx = fmaxf(mx, (float)rp[p]);
#pragma unroll
    for (int msk = 8; msk >= 1; msk >>= 1) mx = fmaxf(mx, __shfl_xor(mx, msk, 32));
    float sum = 0.f;
    for (int p = j; p < Sc; p += 16) {
      float e = __expf((float)rp[p] - mx);
      rp[p] = (_Float16)e;
      sum += e;
    }
#pragma unroll
    for (int msk = 8; msk >= 1; msk >>= 1) sum += __shfl_xor(sum, msk, 32);
    if (j == 0) rsum[row] = sum;
  }
  __syncthreads();

  // Phase 3: O = P * V. wave -> (d-tile = w&3, t-half = w>>2).
  {
    const int d0 = (w & 3) * 16;
    const int thalf = w >> 2;
    v8f o = {};
    for (int c = 0; c < 16; ++c) {
      const int t0 = thalf * 512 + c * 32;
      v16h af = load_a_frag(sc, Sc, lm, t0, lane);          // probs from LDS
      v16h bf = load_bt_frag(vp, Sc, d0 + lm, t0, lane);    // v [d][t]
      o = wmma_f16(af, bf, o);
    }
#pragma unroll
    for (int r = 0; r < 8; ++r)
      obuf[w * 256 + (r + ((lane & 16) ? 8 : 0)) * 16 + lm] = o[r];
  }
  __syncthreads();

  // Phase 4: combine two t-halves, normalize, write hf[b][s][head*64+d].
  const int b = bh >> 3, head = bh & 7;
  for (int e = threadIdx.x; e < 1024; e += 256) {
    int dt = e >> 8, rm = (e >> 4) & 15, n = e & 15;
    float val = obuf[dt * 256 + rm * 16 + n] + obuf[(dt + 4) * 256 + rm * 16 + n];
    val *= 1.0f / rsum[rm];
    size_t idx = (((size_t)b * Sc) + s0 + rm) * Cc + head * HDc + dt * 16 + n;
    hf32[idx] = val;
    hf16[idx] = (_Float16)val;
  }
}

// ---------------------------------------------------------------------------
// INL GEMM: G[r][o] = sum_c hf16[r][c] * inl_w[o][c]  (M=8192, N=512, K=512)
// ---------------------------------------------------------------------------
__global__ void inl_gemm_kernel(const _Float16* __restrict__ hA,
                                const _Float16* __restrict__ Wf,
                                float* __restrict__ G) {
  const int lane = threadIdx.x & 31;
  const int lm = lane & 15;
  const int task = blockIdx.x * 8 + (threadIdx.x >> 5);  // < 256*16 = 4096
  const int m0 = (task >> 4) * 32, n0 = (task & 15) * 32;

  v8f acc[2][2] = {};
  for (int k0 = 0; k0 < Cc; k0 += 32) {
    if (k0 + 32 < Cc) {
      __builtin_prefetch(hA + (size_t)(m0 + lm) * Cc + k0 + 32, 0, 1);
      __builtin_prefetch(Wf + (size_t)(n0 + lm) * Cc + k0 + 32, 0, 1);
    }
    v16h a0 = load_a_frag(hA, Cc, m0 + lm, k0, lane);
    v16h a1 = load_a_frag(hA, Cc, m0 + 16 + lm, k0, lane);
    v16h b0 = load_bt_frag(Wf, Cc, n0 + lm, k0, lane);
    v16h b1 = load_bt_frag(Wf, Cc, n0 + 16 + lm, k0, lane);
    acc[0][0] = wmma_f16(a0, b0, acc[0][0]);
    acc[0][1] = wmma_f16(a0, b1, acc[0][1]);
    acc[1][0] = wmma_f16(a1, b0, acc[1][0]);
    acc[1][1] = wmma_f16(a1, b1, acc[1][1]);
  }
#pragma unroll
  for (int im = 0; im < 2; ++im)
#pragma unroll
    for (int in = 0; in < 2; ++in)
#pragma unroll
      for (int r = 0; r < 8; ++r) {
        int rr = m0 + im * 16 + r + ((lane & 16) ? 8 : 0);
        int oo = n0 + in * 16 + lm;
        G[(size_t)rr * Cc + oo] = acc[im][in][r];
      }
}

// Euler step: hf <- hf + dt * tanh(G + b); refresh f16 copy for next GEMM.
__global__ void inl_update_kernel(float* __restrict__ hf32,
                                  _Float16* __restrict__ hf16,
                                  const float* __restrict__ G,
                                  const float* __restrict__ inl_b) {
  int i = blockIdx.x * blockDim.x + threadIdx.x;
  if (i < Bc * Sc * Cc) {
    float v = hf32[i] + 0.1f * tanhf(G[i] + inl_b[i & (Cc - 1)]);
    hf32[i] = v;
    hf16[i] = (_Float16)v;
  }
}

// ---------------------------------------------------------------------------
// proj GEMM + bias + residual: out[b][o][s] = sum_c hf16[b*S+s][c]*proj_w[o][c]
//                                             + pb[o] + x[b][o][s]
// Tiles staged through LDS so stores along s are coalesced.
// ---------------------------------------------------------------------------
__global__ void proj_gemm_kernel(const _Float16* __restrict__ hA,
                                 const _Float16* __restrict__ Wf,
                                 const float* __restrict__ proj_b,
                                 const float* __restrict__ x,
                                 float* __restrict__ out) {
  __shared__ float stage[8][32][33];
  const int lane = threadIdx.x & 31;
  const int lm = lane & 15;
  const int w = threadIdx.x >> 5;
  const int task = blockIdx.x * 8 + w;                     // < 4096
  const int m0 = (task >> 4) * 32, n0 = (task & 15) * 32;  // m = b*1024+s, n = o

  v8f acc[2][2] = {};
  for (int k0 = 0; k0 < Cc; k0 += 32) {
    if (k0 + 32 < Cc) {
      __builtin_prefetch(hA + (size_t)(m0 + lm) * Cc + k0 + 32, 0, 1);
      __builtin_prefetch(Wf + (size_t)(n0 + lm) * Cc + k0 + 32, 0, 1);
    }
    v16h a0 = load_a_frag(hA, Cc, m0 + lm, k0, lane);
    v16h a1 = load_a_frag(hA, Cc, m0 + 16 + lm, k0, lane);
    v16h b0 = load_bt_frag(Wf, Cc, n0 + lm, k0, lane);
    v16h b1 = load_bt_frag(Wf, Cc, n0 + 16 + lm, k0, lane);
    acc[0][0] = wmma_f16(a0, b0, acc[0][0]);
    acc[0][1] = wmma_f16(a0, b1, acc[0][1]);
    acc[1][0] = wmma_f16(a1, b0, acc[1][0]);
    acc[1][1] = wmma_f16(a1, b1, acc[1][1]);
  }
#pragma unroll
  for (int im = 0; im < 2; ++im)
#pragma unroll
    for (int in = 0; in < 2; ++in)
#pragma unroll
      for (int r = 0; r < 8; ++r)
        stage[w][im * 16 + r + ((lane & 16) ? 8 : 0)][in * 16 + lm] =
            acc[im][in][r];
  __syncthreads();

  const int b = m0 >> 10;
  const int sb = m0 & (Sc - 1);
#pragma unroll 4
  for (int orow = 0; orow < 32; ++orow) {
    int o = n0 + orow;
    size_t idx = (((size_t)b * Cc + o) * Sc) + sb + lane;
    out[idx] = stage[w][lane][orow] + proj_b[o] + x[idx];
  }
}

// ---------------------------------------------------------------------------
// Host-side orchestration (stream-ordered, graph-capture safe).
// ---------------------------------------------------------------------------
extern "C" void kernel_launch(void* const* d_in, const int* in_sizes, int n_in,
                              void* d_out, int out_size, void* d_ws, size_t ws_size,
                              hipStream_t stream) {
  const float* x      = (const float*)d_in[0];
  const float* gns    = (const float*)d_in[1];
  const float* gnb    = (const float*)d_in[2];
  const float* qkv_w  = (const float*)d_in[3];
  const float* qkv_b  = (const float*)d_in[4];
  const float* proj_w = (const float*)d_in[5];
  const float* proj_b = (const float*)d_in[6];
  const float* inl_w  = (const float*)d_in[7];
  const float* inl_b  = (const float*)d_in[8];
  float* out = (float*)d_out;

  // Workspace layout (bytes). Total required: ~50.5 MB.
  char* ws = (char*)d_ws;
  _Float16* qkvw16  = (_Float16*)(ws + 0);          // 1.5 MB  [o][c]
  _Float16* inlw16  = (_Float16*)(ws + 1572864);    // 0.5 MB  [o][c]
  _Float16* projw16 = (_Float16*)(ws + 2097152);    // 0.5 MB  [o][c]
  _Float16* hT16    = (_Float16*)(ws + 2621440);    // 8 MB [b][s][c] (reused as hf16)
  _Float16* q16     = (_Float16*)(ws + 11010048);   // 8 MB [bh][s][d]
  _Float16* k16     = (_Float16*)(ws + 19398656);   // 8 MB [bh][t][d]
  _Float16* v16     = (_Float16*)(ws + 27787264);   // 8 MB [bh][d][t]
  float*    hf32    = (float*)   (ws + 36175872);   // 16 MB [b][s][c]
  float*    G       = (float*)   (ws + 11010048);   // 16 MB, aliases q16+k16 (free post-attn)
  _Float16* hf16    = hT16;                         // hT16 free after qkv GEMM

  prep_w_kernel<<<3072, 256, 0, stream>>>(qkv_w, inl_w, proj_w,
                                          qkvw16, inlw16, projw16);
  gn_kernel<<<64, 256, 0, stream>>>(x, gns, gnb, hT16);
  qkv_gemm_kernel<<<1536, 256, 0, stream>>>(qkvw16, hT16, qkv_b, q16, k16, v16);
  attn_kernel<<<4096, 256, 0, stream>>>(q16, k16, v16, hf32, hf16);
  for (int it = 0; it < 3; ++it) {
    inl_gemm_kernel<<<512, 256, 0, stream>>>(hf16, inlw16, G);
    inl_update_kernel<<<16384, 256, 0, stream>>>(hf32, hf16, G, inl_b);
  }
  proj_gemm_kernel<<<512, 256, 0, stream>>>(hf16, projw16, proj_b, x, out);
}
